// Transformer_4776003633607
// MI455X (gfx1250) — matrix operand order repaired and checked by reading, not compiled, not run on hardware
//
#include <hip/hip_runtime.h>
#include <hip/hip_bf16.h>

// ---------------------------------------------------------------------------
// Problem constants (from reference): B=8, C=1024, L=2048, OC=3*C=3072
// ---------------------------------------------------------------------------
#define BB 8
#define CC 1024
#define LL 2048

typedef __attribute__((ext_vector_type(16))) __bf16          v16bf;
typedef __attribute__((ext_vector_type(16))) unsigned short  v16us;
typedef __attribute__((ext_vector_type(8)))  unsigned short  v8us;
typedef __attribute__((ext_vector_type(8)))  float           v8f;

// ---------------------------------------------------------------------------
// bf16 helpers (bit-level; round-to-nearest-even)
// ---------------------------------------------------------------------------
__device__ __forceinline__ unsigned short f2bf(float f) {
    unsigned u = __float_as_uint(f);
    unsigned r = u + 0x7FFFu + ((u >> 16) & 1u);
    return (unsigned short)(r >> 16);
}

__device__ __forceinline__ v16bf cat8(v8us lo, v8us hi) {
    v16us v = __builtin_shufflevector(lo, hi, 0,1,2,3,4,5,6,7,8,9,10,11,12,13,14,15);
    return __builtin_bit_cast(v16bf, v);
}

// ---------------------------------------------------------------------------
// Direct-from-global WMMA fragment loads (CDNA5 layouts, CDNA5 ISA §7.12.2).
// Every fragment is two contiguous 16-byte runs in memory.
//
// A (MxK, row-major over K):
//   lanes 0-15 : M=lane&15, K = {0..7} and {16..23}   -> p, p+16
//   lanes 16-31: M=lane&15, K = {8..15} and {24..31}  -> p+8, p+24
// ---------------------------------------------------------------------------
__device__ __forceinline__ v16bf frag_a_g(const unsigned short* base,
                                          long rowstride, int m, int lane) {
    int kb0 = (lane >> 4) ? 8 : 0;
    const unsigned short* p = base + (long)m * rowstride + kb0;
    return cat8(*(const v8us*)p, *(const v8us*)(p + 16));
}

// B (KxN), source stored n-major: row n holds K contiguous.
//   lane n needs row n elements K = {kb..kb+7} and {16+kb..16+kb+7}, kb=8*(lane>>4) -> p, p+16
__device__ __forceinline__ v16bf frag_b_g(const unsigned short* base,
                                          long rowstride, int n, int lane) {
    int kh = (lane >> 4) << 3;
    const unsigned short* p = base + (long)n * rowstride + kh;
    return cat8(*(const v8us*)p, *(const v8us*)(p + 16));
}

__device__ __forceinline__ v8f wmma_bf16(v16bf a, v16bf b, v8f c) {
    return __builtin_amdgcn_wmma_f32_16x16x32_bf16(false, a, false, b,
                                                   (short)0, c, false, false);
}

// ---------------------------------------------------------------------------
// Kernel 0a: W f32 -> bf16 (row-major [3C, C]); one-shot, 8 elems/thread
// ---------------------------------------------------------------------------
__global__ void __launch_bounds__(256)
convert_w_kernel(const float* __restrict__ W, unsigned short* __restrict__ Wbf)
{
    long i = ((long)blockIdx.x * 256 + threadIdx.x) * 8;
    const float4* src = (const float4*)(W + i);
    float4 f0 = src[0], f1 = src[1];
    v8us h;
    h[0] = f2bf(f0.x); h[1] = f2bf(f0.y); h[2] = f2bf(f0.z); h[3] = f2bf(f0.w);
    h[4] = f2bf(f1.x); h[5] = f2bf(f1.y); h[6] = f2bf(f1.z); h[7] = f2bf(f1.w);
    *(v8us*)(Wbf + i) = h;
}

// ---------------------------------------------------------------------------
// Kernel 0b: Xt[b,l,c] = bf16(X[b,c,l]); LDS-tiled 32x32 transpose.
// ---------------------------------------------------------------------------
__global__ void __launch_bounds__(256)
transpose_x_kernel(const float* __restrict__ X, unsigned short* __restrict__ Xt)
{
    __shared__ unsigned short tile[32][33];
    const int blk = blockIdx.x;                 // < 8*32*64
    const int b   = blk / (32 * 64);
    const int rem = blk % (32 * 64);
    const int c0  = (rem / 64) * 32;
    const int l0  = (rem % 64) * 32;
    const int tx  = threadIdx.x & 31;
    const int ty  = threadIdx.x >> 5;           // 0..7
#pragma unroll
    for (int q = 0; q < 4; ++q) {
        int c = ty + q * 8;
        tile[c][tx] = f2bf(X[((long)b * CC + c0 + c) * LL + l0 + tx]);
    }
    __syncthreads();
#pragma unroll
    for (int q = 0; q < 4; ++q) {
        int l = ty + q * 8;
        Xt[((long)b * LL + l0 + l) * CC + c0 + tx] = tile[tx][l];
    }
}

// ---------------------------------------------------------------------------
// Kernel 1: qkv[b,o,l] = sum_c W[o,c]*X[b,c,l] + bias[o]
// A = Wbf [3C,C] (row-major over c), B = Xt[b,l,c] (n-major). Wave: 64x32.
// Routes 64-row o-blocks: o<C -> Qt[b,L,C]; <2C -> Kt[b,L,C]; else V[b,C,L]
// ---------------------------------------------------------------------------
__global__ void __launch_bounds__(256)
qkv_gemm_kernel(const unsigned short* __restrict__ Wbf,
                const unsigned short* __restrict__ Xt,
                const float* __restrict__ bias,
                unsigned short* __restrict__ Qt,
                unsigned short* __restrict__ Kt,
                unsigned short* __restrict__ Vb)
{
    const int lane = threadIdx.x & 31;
    const int wid  = threadIdx.x >> 5;
    const int tile = blockIdx.x * 8 + wid;      // < 8*48*64
    const int b    = tile / (48 * 64);
    const int rem  = tile % (48 * 64);
    const int o0   = (rem / 64) * 64;
    const int l0   = (rem % 64) * 32;

    const unsigned short* abase = Wbf + (long)o0 * CC;
    const unsigned short* bbase = Xt + ((long)b * LL + l0) * CC;
    const int ml = lane & 15;

    v8f acc[4][2] = {};
#pragma unroll 2
    for (int c0 = 0; c0 < CC; c0 += 32) {
        __builtin_prefetch(abase + (long)ml * CC + c0 + 32, 0, 0);
        __builtin_prefetch(bbase + (long)ml * CC + c0 + 32, 0, 0);
        v16bf fb0 = frag_b_g(bbase + c0, CC, ml,      lane);
        v16bf fb1 = frag_b_g(bbase + c0, CC, ml + 16, lane);
#pragma unroll
        for (int mt = 0; mt < 4; ++mt) {
            v16bf fa = frag_a_g(abase + c0, CC, ml + mt * 16, lane);
            acc[mt][0] = wmma_bf16(fa, fb0, acc[mt][0]);
            acc[mt][1] = wmma_bf16(fa, fb1, acc[mt][1]);
        }
    }

    const int mb = (lane >> 4) * 8;
#pragma unroll
    for (int mt = 0; mt < 4; ++mt) {
        const int obase = o0 + mt * 16 + mb;
#pragma unroll
        for (int nt = 0; nt < 2; ++nt) {
            const int l = l0 + nt * 16 + ml;
            if (o0 < CC) {              // Q -> Qt[b][l][o] (contiguous in o)
                v8us h;
#pragma unroll
                for (int r = 0; r < 8; ++r)
                    h[r] = f2bf(acc[mt][nt][r] + bias[obase + r]);
                *(v8us*)(Qt + ((long)b * LL + l) * CC + obase) = h;
            } else if (o0 < 2 * CC) {   // K -> Kt[b][l][o-C]
                v8us h;
#pragma unroll
                for (int r = 0; r < 8; ++r)
                    h[r] = f2bf(acc[mt][nt][r] + bias[obase + r]);
                *(v8us*)(Kt + ((long)b * LL + l) * CC + (obase - CC)) = h;
            } else {                    // V -> V[b][o-2C][l] (strided in o)
#pragma unroll
                for (int r = 0; r < 8; ++r)
                    Vb[((long)b * CC + (obase + r - 2 * CC)) * LL + l] =
                        f2bf(acc[mt][nt][r] + bias[obase + r]);
            }
        }
    }
}

// ---------------------------------------------------------------------------
// Kernel 2: St[b,j,i] = (1/sqrt(C)) * sum_c Kt[b,i,c]*Qt[b,j,c]
// Stored transposed (j-major) so the softmax is a row reduction.
// ---------------------------------------------------------------------------
__global__ void __launch_bounds__(256)
attn_scores_kernel(const unsigned short* __restrict__ Kt,
                   const unsigned short* __restrict__ Qt,
                   float* __restrict__ St)
{
    const int lane = threadIdx.x & 31;
    const int wid  = threadIdx.x >> 5;
    const int tile = blockIdx.x * 8 + wid;      // < 8*32*64
    const int b    = tile / (32 * 64);
    const int rem  = tile % (32 * 64);
    const int i0   = (rem / 64) * 64;
    const int j0   = (rem % 64) * 32;

    const unsigned short* abase = Kt + ((long)b * LL + i0) * CC;
    const unsigned short* bbase = Qt + ((long)b * LL + j0) * CC;
    const int ml = lane & 15;

    v8f acc[4][2] = {};
#pragma unroll 2
    for (int c0 = 0; c0 < CC; c0 += 32) {
        __builtin_prefetch(abase + (long)ml * CC + c0 + 32, 0, 0);
        __builtin_prefetch(bbase + (long)ml * CC + c0 + 32, 0, 0);
        v16bf fb0 = frag_b_g(bbase + c0, CC, ml,      lane);
        v16bf fb1 = frag_b_g(bbase + c0, CC, ml + 16, lane);
#pragma unroll
        for (int mt = 0; mt < 4; ++mt) {
            v16bf fa = frag_a_g(abase + c0, CC, ml + mt * 16, lane);
            acc[mt][0] = wmma_bf16(fa, fb0, acc[mt][0]);
            acc[mt][1] = wmma_bf16(fa, fb1, acc[mt][1]);
        }
    }

    const int mb = (lane >> 4) * 8;
    const float scale = 0.03125f;               // 1/sqrt(1024)
#pragma unroll
    for (int mt = 0; mt < 4; ++mt) {
        const int ibase = i0 + mt * 16 + mb;
#pragma unroll
        for (int nt = 0; nt < 2; ++nt) {
            const int j = j0 + nt * 16 + ml;
            float4 s0, s1;
            s0.x = acc[mt][nt][0] * scale; s0.y = acc[mt][nt][1] * scale;
            s0.z = acc[mt][nt][2] * scale; s0.w = acc[mt][nt][3] * scale;
            s1.x = acc[mt][nt][4] * scale; s1.y = acc[mt][nt][5] * scale;
            s1.z = acc[mt][nt][6] * scale; s1.w = acc[mt][nt][7] * scale;
            float4* dst = (float4*)(St + ((long)b * LL + j) * LL + ibase);
            dst[0] = s0; dst[1] = s1;
        }
    }
}

// ---------------------------------------------------------------------------
// Kernel 3: row softmax of St (over i, per (b,j) row). One wave per row,
// shuffle reductions, fully coalesced. Emits bf16 Pt[b,j,i].
// ---------------------------------------------------------------------------
__global__ void __launch_bounds__(256)
softmax_kernel(const float* __restrict__ St, unsigned short* __restrict__ Pt)
{
    const int lane = threadIdx.x & 31;
    const int wid  = threadIdx.x >> 5;
    const int row  = blockIdx.x * 8 + wid;      // < B*L
    const float* src = St + (long)row * LL;

    float m = -3.402823e38f;
    for (int i = lane; i < LL; i += 32) m = fmaxf(m, src[i]);
#pragma unroll
    for (int off = 16; off > 0; off >>= 1) m = fmaxf(m, __shfl_xor(m, off, 32));

    float s = 0.0f;
    for (int i = lane; i < LL; i += 32) s += __expf(src[i] - m);
#pragma unroll
    for (int off = 16; off > 0; off >>= 1) s += __shfl_xor(s, off, 32);

    const float inv = 1.0f / s;
    unsigned short* dst = Pt + (long)row * LL;
    for (int i = lane; i < LL; i += 32)
        dst[i] = f2bf(__expf(src[i] - m) * inv);
}

// ---------------------------------------------------------------------------
// Kernel 4: out[b,c,j] = X[b,c,j] + sum_i V[b,c,i] * Pt[b,j,i]
// A = V[b,C,L] (row-major over i), B = Pt (n-major). Wave: 64x32, K=2048.
// ---------------------------------------------------------------------------
__global__ void __launch_bounds__(256)
out_gemm_kernel(const unsigned short* __restrict__ Vb,
                const unsigned short* __restrict__ Pt,
                const float* __restrict__ X,
                float* __restrict__ Out)
{
    const int lane = threadIdx.x & 31;
    const int wid  = threadIdx.x >> 5;
    const int tile = blockIdx.x * 8 + wid;      // < 8*16*64
    const int b    = tile / (16 * 64);
    const int rem  = tile % (16 * 64);
    const int c0   = (rem / 64) * 64;
    const int j0   = (rem % 64) * 32;

    const unsigned short* abase = Vb + ((long)b * CC + c0) * LL;
    const unsigned short* bbase = Pt + ((long)b * LL + j0) * LL;
    const int ml = lane & 15;

    v8f acc[4][2] = {};
#pragma unroll 2
    for (int k0 = 0; k0 < LL; k0 += 32) {
        __builtin_prefetch(abase + (long)ml * LL + k0 + 32, 0, 0);
        __builtin_prefetch(bbase + (long)ml * LL + k0 + 32, 0, 0);
        v16bf fb0 = frag_b_g(bbase + k0, LL, ml,      lane);
        v16bf fb1 = frag_b_g(bbase + k0, LL, ml + 16, lane);
#pragma unroll
        for (int mt = 0; mt < 4; ++mt) {
            v16bf fa = frag_a_g(abase + k0, LL, ml + mt * 16, lane);
            acc[mt][0] = wmma_bf16(fa, fb0, acc[mt][0]);
            acc[mt][1] = wmma_bf16(fa, fb1, acc[mt][1]);
        }
    }

    const int mb = (lane >> 4) * 8;
#pragma unroll
    for (int mt = 0; mt < 4; ++mt) {
#pragma unroll
        for (int nt = 0; nt < 2; ++nt) {
            const int j = j0 + nt * 16 + ml;
#pragma unroll
            for (int r = 0; r < 8; ++r) {
                const int c = c0 + mt * 16 + mb + r;
                long idx = ((long)b * CC + c) * LL + j;
                Out[idx] = X[idx] + acc[mt][nt][r];
            }
        }
    }
}

// ---------------------------------------------------------------------------
// Launch
// ---------------------------------------------------------------------------
extern "C" void kernel_launch(void* const* d_in, const int* in_sizes, int n_in,
                              void* d_out, int out_size, void* d_ws, size_t ws_size,
                              hipStream_t stream) {
    (void)in_sizes; (void)n_in; (void)out_size; (void)ws_size;

    const float* X    = (const float*)d_in[0];   // [B, C, L]
    const float* W    = (const float*)d_in[1];   // [3C, C]
    const float* bias = (const float*)d_in[2];   // [3C]
    float*       Out  = (float*)d_out;           // [B, C, L]

    char* ws = (char*)d_ws;
    const size_t MB = 1u << 20;
    unsigned short* Qt  = (unsigned short*)(ws);              //  32 MB [B,L,C] bf16
    unsigned short* Kt  = (unsigned short*)(ws +  32 * MB);   //  32 MB [B,L,C] bf16
    unsigned short* Vb  = (unsigned short*)(ws +  64 * MB);   //  32 MB [B,C,L] bf16
    float*          St  = (float*)        (ws +  96 * MB);    // 128 MB [B,L,L] f32 (j-major)
    unsigned short* Pt  = (unsigned short*)(ws + 224 * MB);   //  64 MB [B,L,L] bf16 (j-major)
    unsigned short* Xt  = (unsigned short*)(ws + 288 * MB);   //  32 MB [B,L,C] bf16
    unsigned short* Wbf = (unsigned short*)(ws + 320 * MB);   //   6 MB [3C,C]  bf16

    // K0a: W -> bf16 (3072*1024 / 8 elems per thread)
    convert_w_kernel<<<(3 * CC) * CC / 8 / 256, 256, 0, stream>>>(W, Wbf);

    // K0b: X -> Xt bf16 transpose (32x32 tiles)
    transpose_x_kernel<<<BB * (CC / 32) * (LL / 32), 256, 0, stream>>>(X, Xt);

    // K1: 8 * (3072/64) * (2048/32) = 24576 wave-tiles, 8 waves/block
    qkv_gemm_kernel<<<BB * 48 * 64 / 8, 256, 0, stream>>>(Wbf, Xt, bias, Qt, Kt, Vb);

    // K2: 8 * (2048/64) * (2048/32) = 16384 wave-tiles
    attn_scores_kernel<<<BB * 32 * 64 / 8, 256, 0, stream>>>(Kt, Qt, St);

    // K3: one wave per (b,j) row: B*L waves, 8 per block
    softmax_kernel<<<BB * LL / 8, 256, 0, stream>>>(St, Pt);

    // K4: 8 * (1024/64) * (2048/32) = 8192 wave-tiles
    out_gemm_kernel<<<BB * 16 * 64 / 8, 256, 0, stream>>>(Vb, Pt, X, Out);
}
